// MNet_6064493822008
// MI455X (gfx1250) — matrix operand order repaired and checked by reading, not compiled, not run on hardware
//
#include <hip/hip_runtime.h>

typedef _Float16 v2h  __attribute__((ext_vector_type(2)));
typedef _Float16 v8h  __attribute__((ext_vector_type(8)));
typedef _Float16 v16h __attribute__((ext_vector_type(16)));
typedef float    v4f  __attribute__((ext_vector_type(4)));
typedef float    v8f  __attribute__((ext_vector_type(8)));

#define WROW   72      // padded row stride (halves): 144B -> conflict-free b128 ld
#define NWAVES 4       // 128 threads / wave32
#define TM     32      // rows per wave-tile (2 WMMA sub-tiles of 16)
#define NEUR   64

// A-fragment (16x32 f16): lane<16 -> M=lane, K={0..7,16..23}; lane>=16 -> K={8..15,24..31}
// p = row base + kb*32 + (lane>=16 ? 8 : 0); halves at p[0..7] and p[16..23]
__device__ __forceinline__ v16h ldfragA(const _Float16* p) {
    v8h lo = *(const v8h*)p;
    v8h hi = *(const v8h*)(p + 16);
    return __builtin_shufflevector(lo, hi, 0,1,2,3,4,5,6,7,8,9,10,11,12,13,14,15);
}

// B-fragment (32x16 f16): lane<16 -> N=lane, K={0..7,16..23}; lane>=16 -> N=lane-16, K={8..15,24..31}
// p = W row base + kb*32 + (lane>=16 ? 8 : 0); halves at p[0..7] and p[16..23]
__device__ __forceinline__ v16h ldfragB(const _Float16* p) {
    v8h lo = *(const v8h*)p;
    v8h hi = *(const v8h*)(p + 16);
    return __builtin_shufflevector(lo, hi, 0,1,2,3,4,5,6,7,8,9,10,11,12,13,14,15);
}

__device__ __forceinline__ float silu(float s) {
    return s * __builtin_amdgcn_rcpf(1.0f + __expf(-s));
}

__global__ __launch_bounds__(128) void mlp6_wmma_kernel(
    const float* __restrict__ x,
    const float* __restrict__ W0, const float* __restrict__ b0,
    const float* __restrict__ W1, const float* __restrict__ b1,
    const float* __restrict__ W2, const float* __restrict__ b2,
    const float* __restrict__ W3, const float* __restrict__ b3,
    const float* __restrict__ W4, const float* __restrict__ b4,
    const float* __restrict__ Wout, const float* __restrict__ bout,
    float* __restrict__ out, int n, int nTiles)
{
    __shared__ __align__(16) _Float16 wlds[4][NEUR * WROW];   // W1..W4, f16, [out][in]
    __shared__ __align__(16) _Float16 act[NWAVES][TM * WROW]; // per-wave activation tile
    __shared__ __align__(16) float w0s[NEUR], b0s[NEUR], bls[4][NEUR], wos[NEUR];
    __shared__ float bo_s;

    const int tid = threadIdx.x;

    // ---- one-time weight preload: global f32 -> LDS (f16 for GEMM weights) ----
    {
        const float* Wm[4] = {W1, W2, W3, W4};
        const float* Bm[4] = {b1, b2, b3, b4};
#pragma unroll
        for (int li = 0; li < 4; ++li) {
            const float* Wsrc = Wm[li];
#pragma unroll
            for (int r = 0; r < 32; ++r) {              // 4096 elems / 128 threads
                int idx = r * 128 + tid;
                int nrow = idx >> 6, k = idx & 63;
                wlds[li][nrow * WROW + k] = (_Float16)Wsrc[idx];
            }
            if (tid < NEUR) bls[li][tid] = Bm[li][tid];
        }
        if (tid < NEUR) { w0s[tid] = W0[tid]; b0s[tid] = b0[tid]; wos[tid] = Wout[tid]; }
        if (tid == 0)   bo_s = bout[0];
    }
    __syncthreads();

    const int lane    = tid & 31;
    const int wave    = tid >> 5;
    const int halfSel = lane >> 4;        // 0: lanes 0-15, 1: lanes 16-31
    const int mRow    = lane & 15;
    _Float16* actW = act[wave];

    // ---- hoist per-tile scalars into registers ----
    float biasReg[16];
#pragma unroll
    for (int li = 0; li < 4; ++li)
#pragma unroll
        for (int nt = 0; nt < 4; ++nt)
            biasReg[li * 4 + nt] = bls[li][nt * 16 + mRow];
    const float boR = bo_s;

    for (long t = (long)blockIdx.x * NWAVES + wave; t < nTiles; t += (long)gridDim.x * NWAVES) {
        const long r0 = t * TM;

        // ---- layer 0 (in_f=1): one sample row per lane, vectorized LDS I/O ----
        {
            const long gr = r0 + lane;
            const float xv = (gr < n) ? x[gr] : 0.0f;      // coalesced
#pragma unroll
            for (int j8 = 0; j8 < 8; ++j8) {
                v4f wA = *(const v4f*)&w0s[j8 * 8];        // broadcast (same addr all lanes)
                v4f wB = *(const v4f*)&w0s[j8 * 8 + 4];
                v4f bA = *(const v4f*)&b0s[j8 * 8];
                v4f bB = *(const v4f*)&b0s[j8 * 8 + 4];
                v8h o;
#pragma unroll
                for (int k = 0; k < 4; ++k) {
                    o[k]     = (_Float16)silu(xv * wA[k] + bA[k]);
                    o[k + 4] = (_Float16)silu(xv * wB[k] + bB[k]);
                }
                *(v8h*)&actW[lane * WROW + j8 * 8] = o;    // ds_store_b128
            }
        }

        // ---- layers 1..4: 32x64 @ 64x64, two independent 16-row WMMA sub-tiles ----
#pragma unroll
        for (int li = 0; li < 4; ++li) {
            const _Float16* wl = wlds[li];
            v16h a00 = ldfragA(actW + mRow * WROW +        halfSel * 8);       // sub0, K0
            v16h a01 = ldfragA(actW + mRow * WROW + 32   + halfSel * 8);       // sub0, K1
            v16h a10 = ldfragA(actW + (16 + mRow) * WROW +      halfSel * 8);  // sub1, K0
            v16h a11 = ldfragA(actW + (16 + mRow) * WROW + 32 + halfSel * 8);  // sub1, K1
#pragma unroll
            for (int nt = 0; nt < 4; ++nt) {
                const int n0 = nt * 16;
                const _Float16* wp = wl + (n0 + mRow) * WROW + halfSel * 8;
                v16h bf0 = ldfragB(wp);        // K block 0
                v16h bf1 = ldfragB(wp + 32);   // K block 1
                const float bias = biasReg[li * 4 + nt];
                v8f acc0 = {bias, bias, bias, bias, bias, bias, bias, bias};
                v8f acc1 = acc0;
                // two independent accumulator chains interleave in the hazard slots
                acc0 = __builtin_amdgcn_wmma_f32_16x16x32_f16(false, a00, false, bf0, (short)0, acc0, false, false);
                acc1 = __builtin_amdgcn_wmma_f32_16x16x32_f16(false, a10, false, bf0, (short)0, acc1, false, false);
                acc0 = __builtin_amdgcn_wmma_f32_16x16x32_f16(false, a01, false, bf1, (short)0, acc0, false, false);
                acc1 = __builtin_amdgcn_wmma_f32_16x16x32_f16(false, a11, false, bf1, (short)0, acc1, false, false);
                // SiLU + f16 store back (in place; A-fragments already in registers).
                // D layout: VGPR v -> row v + 8*halfSel, col = lane&15.
#pragma unroll
                for (int v = 0; v < 8; ++v) {
                    actW[(v + 8 * halfSel) * WROW + n0 + mRow]        = (_Float16)silu(acc0[v]);
                    actW[(16 + v + 8 * halfSel) * WROW + n0 + mRow]   = (_Float16)silu(acc1[v]);
                }
            }
        }

        // ---- output layer (64 -> 1): one row per lane, v8h act loads, fma_mix dot ----
        {
            float sum = boR;
#pragma unroll
            for (int c8 = 0; c8 < 8; ++c8) {
                v8h av = *(const v8h*)&actW[lane * WROW + c8 * 8];   // ds_load_b128
                v4f wA = *(const v4f*)&wos[c8 * 8];                  // broadcast
                v4f wB = *(const v4f*)&wos[c8 * 8 + 4];
#pragma unroll
                for (int k = 0; k < 4; ++k) {
                    sum += (float)av[k]     * wA[k];                 // -> v_fma_mix_f32
                    sum += (float)av[k + 4] * wB[k];
                }
            }
            const long gr = r0 + lane;
            if (gr < n) out[gr] = sum;                               // coalesced
        }
    }
}

extern "C" void kernel_launch(void* const* d_in, const int* in_sizes, int n_in,
                              void* d_out, int out_size, void* d_ws, size_t ws_size,
                              hipStream_t stream) {
    const float* x    = (const float*)d_in[0];
    const float* W0   = (const float*)d_in[1];
    const float* b0   = (const float*)d_in[2];
    const float* W1   = (const float*)d_in[3];
    const float* b1   = (const float*)d_in[4];
    const float* W2   = (const float*)d_in[5];
    const float* b2   = (const float*)d_in[6];
    const float* W3   = (const float*)d_in[7];
    const float* b3   = (const float*)d_in[8];
    const float* W4   = (const float*)d_in[9];
    const float* b4   = (const float*)d_in[10];
    const float* Wout = (const float*)d_in[11];
    const float* bout = (const float*)d_in[12];

    const int n = in_sizes[0];              // 2097152 rows
    const int nTiles = (n + TM - 1) / TM;   // 32-row tiles, one per wave iteration
    const int grid = 4096;                  // persistent blocks; grid-stride over tiles

    mlp6_wmma_kernel<<<grid, 128, 0, stream>>>(
        x, W0, b0, W1, b1, W2, b2, W3, b3, W4, b4, Wout, bout,
        (float*)d_out, n, nTiles);
}
